// CrossAttentionInjector_523986010280
// MI455X (gfx1250) — matrix operand order repaired and checked by reading, not compiled, not run on hardware
//
#include <hip/hip_runtime.h>

// ---------------------------------------------------------------------------
// CDNA5 (gfx1250) wave32 WMMA implementation of CrossAttentionInjector.
// All GEMMs use v_wmma_f32_16x16x32_bf16 (fp32 data converted to bf16 in
// registers, fp32 accumulation). Column strides are template constants so
// strided fragment loads become base + immediate-offset vmem ops; batch/head
// de-interleave uses shift/mask (no integer division).
// ---------------------------------------------------------------------------

typedef __attribute__((ext_vector_type(16))) __bf16 v16bf;
typedef __attribute__((ext_vector_type(8)))  float  v8f;

__device__ __forceinline__ __bf16 f2bf(float x) { return (__bf16)x; }

// Batch offset: (z>>sh)*outer + (z & ((1<<sh)-1))*inner.  sh=0 -> z*outer.
__device__ __forceinline__ long long boff(int z, int sh, long long o, long long i) {
  return (long long)(z >> sh) * o + (long long)(z & ((1 << sh) - 1)) * i;
}

// A-fragment (16x32 bf16, MxK). Contiguous K order (matches B):
//   lanes 0-15: M=lane,  K = 0..15; lanes16-31: M=lane-16, K=16..31
// element j: K = kt + h16 + j.  Base pointer already folds
// (kt + h16), so per-element offsets are compile-time constants.
template<int CS>
__device__ __forceinline__ v16bf load_a_frag(const float* __restrict__ rowp,
                                             int kt, int h8) {
  v16bf f;
  if constexpr (CS == 1) {
    const float* p = rowp + kt + h8;
    const float4* p0 = reinterpret_cast<const float4*>(p);
    const float4* p1 = reinterpret_cast<const float4*>(p + 16);
    float4 u0 = p0[0], u1 = p0[1], u2 = p0[2], u3 = p0[3];
    f[0]=f2bf(u0.x);  f[1]=f2bf(u0.y);  f[2]=f2bf(u0.z);  f[3]=f2bf(u0.w);
    f[4]=f2bf(u1.x);  f[5]=f2bf(u1.y);  f[6]=f2bf(u1.z);  f[7]=f2bf(u1.w);
    f[8]=f2bf(u2.x);  f[9]=f2bf(u2.y);  f[10]=f2bf(u2.z); f[11]=f2bf(u2.w);
    f[12]=f2bf(u3.x); f[13]=f2bf(u3.y); f[14]=f2bf(u3.z); f[15]=f2bf(u3.w);
  } else {
    const float* p = rowp + (long long)(kt + h8) * CS;
#pragma unroll
    for (int j = 0; j < 16; ++j) {
      const int koff = j;   // compile-time
      f[j] = f2bf(p[koff * CS]);                        // immediate offset
    }
  }
  return f;
}

// B-fragment (32x16 bf16, KxN). lanes 0-15: N=lane, K=0..15; lanes16-31:
// N=lane-16, K=16..31. element j: K = kt + h16 + j.
template<int CS>
__device__ __forceinline__ v16bf load_b_frag(const float* __restrict__ colp,
                                             int kt, int h16) {
  v16bf f;
  if constexpr (CS == 1) {
    const float4* p = reinterpret_cast<const float4*>(colp + kt + h16);
    float4 u0 = p[0], u1 = p[1], u2 = p[2], u3 = p[3];
    f[0]=f2bf(u0.x);  f[1]=f2bf(u0.y);  f[2]=f2bf(u0.z);  f[3]=f2bf(u0.w);
    f[4]=f2bf(u1.x);  f[5]=f2bf(u1.y);  f[6]=f2bf(u1.z);  f[7]=f2bf(u1.w);
    f[8]=f2bf(u2.x);  f[9]=f2bf(u2.y);  f[10]=f2bf(u2.z); f[11]=f2bf(u2.w);
    f[12]=f2bf(u3.x); f[13]=f2bf(u3.y); f[14]=f2bf(u3.z); f[15]=f2bf(u3.w);
  } else {
    const float* p = colp + (long long)(kt + h16) * CS;
#pragma unroll
    for (int j = 0; j < 16; ++j)
      f[j] = f2bf(p[j * CS]);                           // immediate offset
  }
  return f;
}

// C/D layout: VGPR r, lanes0-15: M=r, N=lane; lanes16-31: M=r+8, N=lane-16.
__device__ __forceinline__ void store_tile(float* __restrict__ D, int rsD, int csD,
                                           const v8f& acc, int mBase, int nBase,
                                           int lane, const float* __restrict__ bias,
                                           const float* __restrict__ cb, float scale) {
  const int n = nBase + (lane & 15);
  const int h8 = (lane >> 4) << 3;
  float addv = (bias ? bias[n] : 0.f) + (cb ? cb[n] : 0.f);
#pragma unroll
  for (int r = 0; r < 8; ++r) {
    int m = mBase + h8 + r;
    D[(long long)m * rsD + (long long)n * csD] = acc[r] * scale + addv;
  }
}

// Generic batched GEMM: D = scale * A @ B' + bias + colBias.
// A element (m,k) at m*rsA + k*CSA; B' element (k,n) at n*rsB + k*CSB.
// Block: 256 thr = 8 waves (4 along M x 2 along N); wave tile 32x32 (2x2 WMMA);
// block tile 128x64. All problem dims divide evenly -> no bounds checks.
template<int CSA, int CSB>
__global__ __launch_bounds__(256)
void gemm_wmma_kernel(const float* __restrict__ A, int shA, long long oA, long long iA, int rsA,
                      const float* __restrict__ Bm, int shB, long long oB, long long iB, int rsB,
                      const float* __restrict__ bias,
                      const float* __restrict__ colBias, int shCB, int nCB,
                      float* __restrict__ D, int shD, long long oD, long long iD, int rsD, int csD,
                      int Kdim, float scale) {
  const int z = blockIdx.z;
  A  += boff(z, shA, oA, iA);
  Bm += boff(z, shB, oB, iB);
  D  += boff(z, shD, oD, iD);
  const float* cb = colBias ? (colBias + (long long)(z >> shCB) * nCB) : nullptr;

  const int lane  = threadIdx.x & 31;
  const int wv    = threadIdx.x >> 5;
  const int mWave = blockIdx.y * 128 + (wv & 3) * 32;
  const int nWave = blockIdx.x * 64  + (wv >> 2) * 32;

  const int h8   = (lane >> 4) << 3;
  const int h16  = (lane >> 4) << 4;
  const int rsel = lane & 15;

  const float* aRow0 = A  + (long long)(mWave + rsel)      * rsA;
  const float* aRow1 = A  + (long long)(mWave + 16 + rsel) * rsA;
  const float* bCol0 = Bm + (long long)(nWave + rsel)      * rsB;
  const float* bCol1 = Bm + (long long)(nWave + 16 + rsel) * rsB;

  v8f acc00 = {0.f,0.f,0.f,0.f,0.f,0.f,0.f,0.f};
  v8f acc01 = acc00, acc10 = acc00, acc11 = acc00;

  for (int kt = 0; kt < Kdim; kt += 32) {
    v16bf a0 = load_a_frag<CSA>(aRow0, kt, h16);
    v16bf a1 = load_a_frag<CSA>(aRow1, kt, h16);
    v16bf b0 = load_b_frag<CSB>(bCol0, kt, h16);
    v16bf b1 = load_b_frag<CSB>(bCol1, kt, h16);
    // speculative prefetch of next K tile (global_prefetch_b8); harmless at end
    if constexpr (CSA == 1) {
      __builtin_prefetch(aRow0 + kt + 32 + h8, 0, 3);
      __builtin_prefetch(aRow1 + kt + 32 + h8, 0, 3);
    }
    if constexpr (CSB == 1) {
      __builtin_prefetch(bCol0 + kt + 32 + h16, 0, 3);
      __builtin_prefetch(bCol1 + kt + 32 + h16, 0, 3);
    }
    acc00 = __builtin_amdgcn_wmma_f32_16x16x32_bf16(false, a0, false, b0, (short)0, acc00, false, false);
    acc01 = __builtin_amdgcn_wmma_f32_16x16x32_bf16(false, a0, false, b1, (short)0, acc01, false, false);
    acc10 = __builtin_amdgcn_wmma_f32_16x16x32_bf16(false, a1, false, b0, (short)0, acc10, false, false);
    acc11 = __builtin_amdgcn_wmma_f32_16x16x32_bf16(false, a1, false, b1, (short)0, acc11, false, false);
  }

  store_tile(D, rsD, csD, acc00, mWave,      nWave,      lane, bias, cb, scale);
  store_tile(D, rsD, csD, acc01, mWave,      nWave + 16, lane, bias, cb, scale);
  store_tile(D, rsD, csD, acc10, mWave + 16, nWave,      lane, bias, cb, scale);
  store_tile(D, rsD, csD, acc11, mWave + 16, nWave + 16, lane, bias, cb, scale);
}

// L2-normalize rows of length 64 (cond_proj). One wave per row, float2/lane.
__global__ __launch_bounds__(256) void normalize64(float* __restrict__ data) {
  const int lane = threadIdx.x & 31;
  const long long row = (long long)blockIdx.x * 8 + (threadIdx.x >> 5);
  float2* p = reinterpret_cast<float2*>(data + row * 64);
  float2 v = p[lane];
  float ss = v.x * v.x + v.y * v.y;
#pragma unroll
  for (int off = 16; off > 0; off >>= 1) ss += __shfl_xor(ss, off);
  float inv = 1.f / fmaxf(sqrtf(ss), 1e-12f);
  v.x *= inv; v.y *= inv;
  p[lane] = v;
}

// centrality[b,n] = v_n . (sum_m v_m) - ||v_n||^2   (off-diagonal Smat rowsum)
__global__ __launch_bounds__(64) void centrality_kernel(const float* __restrict__ cproj,
                                                        float* __restrict__ cent) {
  __shared__ float T[64];
  const int b = blockIdx.x, d = threadIdx.x;
  const float* cp = cproj + (long long)b * 512 * 64;
  float t = 0.f;
  for (int n = 0; n < 512; ++n) t += cp[n * 64 + d];
  T[d] = t;
  __syncthreads();
  for (int n = d; n < 512; n += 64) {
    const float* v = cp + n * 64;
    float s = 0.f;
#pragma unroll 8
    for (int dd = 0; dd < 64; ++dd) s += v[dd] * (T[dd] - v[dd]);
    cent[b * 512 + n] = s;
  }
}

// top-k via rank counting (ties broken by index, matching jax top_k), mask,
// argmax fallback for all-false rows; emits additive column bias 0 / -1e9.
__global__ __launch_bounds__(512) void select_topk(const float* __restrict__ cent,
                                                   const unsigned char* __restrict__ mask,
                                                   float* __restrict__ cbias) {
  __shared__ float s[512];
  __shared__ int cnt;
  const int b = blockIdx.x, n = threadIdx.x;
  s[n] = cent[b * 512 + n];
  if (n == 0) cnt = 0;
  __syncthreads();
  const float sn = s[n];
  int rank = 0;
  for (int m = 0; m < 512; ++m) {
    float sm = s[m];
    rank += (sm > sn) || (sm == sn && m < n);
  }
  bool allowed = (rank < 307) && (mask[b * 512 + n] != 0);
  if (allowed) atomicAdd(&cnt, 1);
  __syncthreads();
  if (cnt == 0 && rank == 0) allowed = true;  // top-1 fallback
  cbias[b * 512 + n] = allowed ? 0.f : -1e9f;
}

// softmax over rows of length 512 (mask bias already added). One wave per row,
// float4 per lane x4 -> coalesced b128 traffic.
__global__ __launch_bounds__(256) void softmax512(float* __restrict__ data) {
  const int lane = threadIdx.x & 31;
  const long long row = (long long)blockIdx.x * 8 + (threadIdx.x >> 5);
  float4* p = reinterpret_cast<float4*>(data + row * 512);
  float4 v[4];
  float m = -3.4e38f;
#pragma unroll
  for (int i = 0; i < 4; ++i) {
    v[i] = p[lane + 32 * i];
    m = fmaxf(m, fmaxf(fmaxf(v[i].x, v[i].y), fmaxf(v[i].z, v[i].w)));
  }
#pragma unroll
  for (int off = 16; off > 0; off >>= 1) m = fmaxf(m, __shfl_xor(m, off));
  float s = 0.f;
#pragma unroll
  for (int i = 0; i < 4; ++i) {
    v[i].x = __expf(v[i].x - m); v[i].y = __expf(v[i].y - m);
    v[i].z = __expf(v[i].z - m); v[i].w = __expf(v[i].w - m);
    s += v[i].x + v[i].y + v[i].z + v[i].w;
  }
#pragma unroll
  for (int off = 16; off > 0; off >>= 1) s += __shfl_xor(s, off);
  float inv = 1.f / s;
#pragma unroll
  for (int i = 0; i < 4; ++i) {
    v[i].x *= inv; v[i].y *= inv; v[i].z *= inv; v[i].w *= inv;
    p[lane + 32 * i] = v[i];
  }
}

extern "C" void kernel_launch(void* const* d_in, const int* in_sizes, int n_in,
                              void* d_out, int out_size, void* d_ws, size_t ws_size,
                              hipStream_t stream) {
  (void)in_sizes; (void)n_in; (void)out_size; (void)ws_size;
  const float* h    = (const float*)d_in[0];        // (8,256,32,32)
  const float* cf   = (const float*)d_in[1];        // (8,512,512)
  const unsigned char* mask = (const unsigned char*)d_in[2];  // (8,512) bool
  const float* Wq = (const float*)d_in[3];  const float* bq = (const float*)d_in[4];
  const float* Wk = (const float*)d_in[5];  const float* bk = (const float*)d_in[6];
  const float* Wv = (const float*)d_in[7];  const float* bv = (const float*)d_in[8];
  // d_in[9]/d_in[10] = Wrq/brq: provably unused in the output -> skipped.
  const float* Wrk = (const float*)d_in[11]; const float* brk = (const float*)d_in[12];
  const float* Wo  = (const float*)d_in[13]; const float* bo  = (const float*)d_in[14];
  float* out = (float*)d_out;

  float* ws    = (float*)d_ws;
  float* qbuf  = ws;                  // 8*1024*512
  float* kbuf  = qbuf  + 4194304;     // 8*512*512
  float* vbuf  = kbuf  + 2097152;     // 8*512*512
  float* cproj = vbuf  + 2097152;     // 8*512*64
  float* cent  = cproj + 262144;      // 8*512
  float* cbias = cent  + 4096;        // 8*512
  float* probs = cbias + 4096;        // 64*1024*512 (scores -> softmax in place)
  float* attnb = probs + 33554432;    // 8*1024*512

  dim3 blk(256);

  // 1) Q = h^T(B,S,C) @ Wq^T + bq -> qbuf (B,S,512). A: (m,k)->h[b,k*S+m]
  gemm_wmma_kernel<1024, 1><<<dim3(8, 8, 8), blk, 0, stream>>>(
      h,   0, 262144LL, 0LL, 1,
      Wq,  0, 0LL, 0LL, 256,
      bq, nullptr, 0, 0,
      qbuf, 0, 524288LL, 0LL, 512, 1,
      256, 1.0f);

  // 2) K = cond_feats @ Wk^T + bk -> kbuf (B,512,512)
  gemm_wmma_kernel<1, 1><<<dim3(8, 4, 8), blk, 0, stream>>>(
      cf,  0, 262144LL, 0LL, 512,
      Wk,  0, 0LL, 0LL, 512,
      bk, nullptr, 0, 0,
      kbuf, 0, 262144LL, 0LL, 512, 1,
      512, 1.0f);

  // 3) V = cond_feats @ Wv^T + bv -> vbuf
  gemm_wmma_kernel<1, 1><<<dim3(8, 4, 8), blk, 0, stream>>>(
      cf,  0, 262144LL, 0LL, 512,
      Wv,  0, 0LL, 0LL, 512,
      bv, nullptr, 0, 0,
      vbuf, 0, 262144LL, 0LL, 512, 1,
      512, 1.0f);

  // 4) cond_proj = cond_feats @ Wrk^T + brk -> cproj (B,512,64), then normalize
  gemm_wmma_kernel<1, 1><<<dim3(1, 4, 8), blk, 0, stream>>>(
      cf,  0, 262144LL, 0LL, 512,
      Wrk, 0, 0LL, 0LL, 512,
      brk, nullptr, 0, 0,
      cproj, 0, 32768LL, 0LL, 64, 1,
      512, 1.0f);
  normalize64<<<512, 256, 0, stream>>>(cproj);

  // 5) centrality + top-k mask -> additive column bias (B,512)
  centrality_kernel<<<8, 64, 0, stream>>>(cproj, cent);
  select_topk<<<8, 512, 0, stream>>>(cent, mask, cbias);

  // 6) scores = Q @ K^T / 8 + maskbias -> probs (B*8, 1024, 512); z = b*8+h
  gemm_wmma_kernel<1, 1><<<dim3(8, 8, 64), blk, 0, stream>>>(
      qbuf, 3, 524288LL, 64LL, 512,
      kbuf, 3, 262144LL, 64LL, 512,
      nullptr, cbias, 3, 512,
      probs, 0, 524288LL, 0LL, 512, 1,
      64, 0.125f);

  // 7) softmax over keys (row length 512), in place
  softmax512<<<8192, 256, 0, stream>>>(probs);

  // 8) attn_out = P @ V -> attnb (B,S,512) head-interleaved (NN GEMM on B)
  gemm_wmma_kernel<1, 512><<<dim3(1, 8, 64), blk, 0, stream>>>(
      probs, 0, 524288LL, 0LL, 512,
      vbuf,  3, 262144LL, 64LL, 1,
      nullptr, nullptr, 0, 0,
      attnb, 3, 524288LL, 64LL, 512, 1,
      512, 1.0f);

  // 9) out = attn_out @ Wo^T + bo, stored transposed to (B,C,H,W)
  gemm_wmma_kernel<1, 1><<<dim3(4, 8, 8), blk, 0, stream>>>(
      attnb, 0, 524288LL, 0LL, 512,
      Wo,    0, 0LL, 0LL, 512,
      bo, nullptr, 0, 0,
      out, 0, 262144LL, 0LL, 1, 1024,
      512, 1.0f);
}
